// MultiheadAttention_41841571398153
// MI455X (gfx1250) — matrix operand order repaired and checked by reading, not compiled, not run on hardware
//
#include <hip/hip_runtime.h>
#include <hip/hip_bf16.h>
#include <stdint.h>

typedef __bf16 bf16;
typedef __attribute__((ext_vector_type(16))) __bf16 v16bf;
typedef __attribute__((ext_vector_type(8)))  float   v8f;
typedef __attribute__((ext_vector_type(4)))  unsigned int v4u;
typedef __attribute__((ext_vector_type(8)))  int v8i;
typedef __attribute__((ext_vector_type(4)))  int v4i;

#define DTOK 1024
#define NH   16
#define DH   64
#define BATCH 4
#define SEQ  2048
#define BSROWS (BATCH*SEQ)   /* 8192 */

#if defined(__gfx1250__) && __has_builtin(__builtin_amdgcn_tensor_load_to_lds) && \
    __has_builtin(__builtin_amdgcn_s_wait_tensorcnt)
#define USE_TDM 1
#else
#define USE_TDM 0
#endif

union Frag16 { v16bf v; uint4 q[2]; };

static __device__ __forceinline__ v8f wmma_bf16(v16bf a, v16bf b, v8f c) {
  return __builtin_amdgcn_wmma_f32_16x16x32_bf16(false, a, false, b, (short)0, c, false, false);
}

// ---------------- conversion kernels ----------------

__global__ __launch_bounds__(256) void k_cvt_x(const float* __restrict__ x,
                                               bf16* __restrict__ xb, int n) {
  int i = (blockIdx.x * 256 + threadIdx.x) * 4;
  if (i >= n) return;
  float4 f = *(const float4*)(x + i);
  union { bf16 b[4]; uint2 u; } pk;
  pk.b[0] = (bf16)f.x; pk.b[1] = (bf16)f.y; pk.b[2] = (bf16)f.z; pk.b[3] = (bf16)f.w;
  *(uint2*)(xb + i) = pk.u;
}

// Wt[z][n][k] = bf16(W[z][k][n]) via LDS 32x32 tile transpose (coalesced both sides)
__global__ __launch_bounds__(256) void k_cvt_wt(const float* __restrict__ Wq,
                                                const float* __restrict__ Wk,
                                                const float* __restrict__ Wv,
                                                const float* __restrict__ Wo,
                                                bf16* __restrict__ wt) {
  __shared__ float tile[32][33];
  const int z = blockIdx.z;
  const float* W = (z == 0) ? Wq : (z == 1) ? Wk : (z == 2) ? Wv : Wo;
  bf16* out = wt + (size_t)z * DTOK * DTOK;
  const int kt = blockIdx.x * 32, nt = blockIdx.y * 32;
  const int tx = threadIdx.x & 31, ty = threadIdx.x >> 5;   // 32 x 8
#pragma unroll
  for (int j = 0; j < 4; ++j)
    tile[ty + j * 8][tx] = W[(size_t)(kt + ty + j * 8) * DTOK + nt + tx];
  __syncthreads();
#pragma unroll
  for (int j = 0; j < 4; ++j)
    out[(size_t)(nt + ty + j * 8) * DTOK + kt + tx] = (bf16)tile[tx][ty + j * 8];
}

// ---------------- TDM: DMA a 64(N) x 32(K) bf16 tile of Wt into LDS ----------------
#if USE_TDM
static __device__ __forceinline__ void tdm_load_tile(unsigned lds_off, const bf16* gaddr) {
  unsigned long long ga = (unsigned long long)(uintptr_t)gaddr;
  v4u g0;
  g0[0] = 1u;                                            // count=1, user descriptor
  g0[1] = lds_off;                                       // lds_addr (bytes)
  g0[2] = (unsigned)(ga & 0xffffffffu);                  // global_addr[31:0]
  g0[3] = (unsigned)((ga >> 32) & 0x01ffffffu) | (2u << 30); // global_addr[56:32] | type=2
  v8i g1;
  g1[0] = 0x00010000;          // workgroup_mask=0, data_size=1 (2 bytes/elt)
  g1[1] = (int)(1024u << 16);  // tensor_dim0 = 1024 (bits 79:48, low half)
  g1[2] = (int)(1024u << 16);  // tensor_dim0 hi=0 ; tensor_dim1 = 1024 (bits 111:80, low half)
  g1[3] = (int)(32u << 16);    // tensor_dim1 hi=0 ; tile_dim0 = 32 (bits 127:112)
  g1[4] = 64;                  // tile_dim1 = 64 ; tile_dim2 = 0
  g1[5] = 1024;                // tensor_dim0_stride = 1024 elements
  g1[6] = 0;
  g1[7] = 0;
  v4i z4 = {0, 0, 0, 0};
  v8i z8 = {0, 0, 0, 0, 0, 0, 0, 0};
  __builtin_amdgcn_tensor_load_to_lds(g0, g1, z4, z4, z8, 0);
}
#endif

// ---------------- GEMM cores: one wave -> 16(M) x 64(N) tile ----------------
// A row-major [M][K]; Bt = B^T [N][K] (column N contiguous in K). K must be pow2.

static __device__ __forceinline__ void mm_acc_db(const bf16* __restrict__ A,
                                                 const bf16* __restrict__ Bt,
                                                 int row0, int n0, int K, v8f acc[4]) {
  const int lane = threadIdx.x & 31;
  const int half = lane >> 4;
  const int r    = lane & 15;
  const bf16* ap  = A  + (size_t)(row0 + r) * K + half * 8;
  const bf16* bp0 = Bt + (size_t)(n0  + r) * K + half * 8;
  Frag16 a[2], b[2][4];
  auto LOAD = [&](int buf, int k) {
    a[buf].q[0] = *(const uint4*)(ap + k);
    a[buf].q[1] = *(const uint4*)(ap + k + 16);
#pragma unroll
    for (int t = 0; t < 4; ++t) {
      const bf16* bp = bp0 + (size_t)t * 16 * K + k;
      b[buf][t].q[0] = *(const uint4*)(bp);
      b[buf][t].q[1] = *(const uint4*)(bp + 16);
    }
  };
  LOAD(0, 0);
  for (int k = 0; k < K; k += 64) {
    LOAD(1, k + 32);
#pragma unroll
    for (int t = 0; t < 4; ++t) acc[t] = wmma_bf16(a[0].v, b[0][t].v, acc[t]);
    LOAD(0, (k + 64) & (K - 1));              // last iter reloads k=0 (unused)
#pragma unroll
    for (int t = 0; t < 4; ++t) acc[t] = wmma_bf16(a[1].v, b[1][t].v, acc[t]);
  }
}

#if USE_TDM
// B tiles staged in LDS by the Tensor Data Mover (wave 0 issues DMA), A double-buffered in regs.
static __device__ __forceinline__ void mm_acc_tdm(const bf16* __restrict__ A,
                                                  const bf16* __restrict__ Btile, // Bt + n0*K
                                                  int row0, int K, v8f acc[4],
                                                  bf16* sB /* [2][64*32] */) {
  const int lane = threadIdx.x & 31;
  const int half = lane >> 4;
  const int r    = lane & 15;
  const unsigned lds0 = (unsigned)(uintptr_t)sB;   // low 32 bits of flat shared addr = LDS offset
  const bf16* ap = A + (size_t)(row0 + r) * K + half * 8;

  if (threadIdx.x < 32) tdm_load_tile(lds0, Btile);
  Frag16 a0, a1;
  a0.q[0] = *(const uint4*)(ap);
  a0.q[1] = *(const uint4*)(ap + 16);

  const int nsteps = K / 32;
  for (int i = 0; i < nsteps; ++i) {
    const int kn = ((i + 1) * 32) & (K - 1);     // last iter: dummy prefetch of k=0
    if (threadIdx.x < 32)
      tdm_load_tile(lds0 + (unsigned)(((i + 1) & 1) * (64 * 32 * 2)), Btile + kn);
    a1.q[0] = *(const uint4*)(ap + kn);
    a1.q[1] = *(const uint4*)(ap + kn + 16);
    if (threadIdx.x < 32) __builtin_amdgcn_s_wait_tensorcnt(1);  // tile i is resident
    __syncthreads();
    const bf16* lb = sB + (i & 1) * (64 * 32);
#pragma unroll
    for (int t = 0; t < 4; ++t) {
      Frag16 b;
      const bf16* bp = lb + (t * 16 + r) * 32 + half * 8;
      b.q[0] = *(const uint4*)(bp);
      b.q[1] = *(const uint4*)(bp + 16);
      acc[t] = wmma_bf16(a0.v, b.v, acc[t]);
    }
    __syncthreads();                             // tile i fully consumed before DMA overwrites it
    a0 = a1;
  }
}
#endif

// ---------------- fused QKV projection ----------------
__global__ __launch_bounds__(128) void k_qkv(const bf16* __restrict__ xb,
                                             const bf16* __restrict__ wt,
                                             const float* __restrict__ bq,
                                             const float* __restrict__ bk,
                                             const float* __restrict__ bv,
                                             bf16* __restrict__ Qb,
                                             bf16* __restrict__ Kb,
                                             bf16* __restrict__ Vt) {
  const int wave = threadIdx.x >> 5;
  const int row0 = (blockIdx.x * 4 + wave) * 16;
  const int n0   = blockIdx.y * 64;
  const int w    = blockIdx.z;
  const bf16* Bt = wt + (size_t)w * DTOK * DTOK;
  const float* bias = (w == 0) ? bq : (w == 1) ? bk : bv;

  v8f acc[4] = {};
#if USE_TDM
  __shared__ __align__(16) bf16 sB[2][64 * 32];
  mm_acc_tdm(xb, Bt + (size_t)n0 * DTOK, row0, DTOK, acc, &sB[0][0]);
#else
  mm_acc_db(xb, Bt, row0, n0, DTOK, acc);
#endif

  const int lane = threadIdx.x & 31;
  const int half = lane >> 4;
  const int r    = lane & 15;
  const int b  = row0 >> 11;
  const int s0 = row0 & (SEQ - 1);
#pragma unroll
  for (int t = 0; t < 4; ++t) {
    const int n = n0 + t * 16 + r;
    const int h = n >> 6, d = n & 63;
    const float bi = bias[n];
    if (w < 2) {
      bf16* out = ((w == 0) ? Qb : Kb) + ((size_t)(b * NH + h) * SEQ) * DH + d;
#pragma unroll
      for (int j = 0; j < 8; ++j) {
        const int s = s0 + j + half * 8;
        out[(size_t)s * DH] = (bf16)(acc[t][j] + bi);
      }
    } else {
      union { bf16 e[8]; uint4 u; } pk;
#pragma unroll
      for (int j = 0; j < 8; ++j) pk.e[j] = (bf16)(acc[t][j] + bi);
      *(uint4*)(Vt + ((size_t)(b * NH + h) * DH + d) * SEQ + s0 + half * 8) = pk.u;
    }
  }
}

// ---------------- flash attention per (b,h) ----------------
__global__ __launch_bounds__(128) void k_attn(const bf16* __restrict__ Qb,
                                              const bf16* __restrict__ Kb,
                                              const bf16* __restrict__ Vt,
                                              bf16* __restrict__ Ctx) {
  __shared__ __align__(16) bf16 sP[4][16 * 32];
  const int wave = threadIdx.x >> 5;
  const int lane = threadIdx.x & 31;
  const int half = lane >> 4;
  const int r    = lane & 15;
  const int bh   = blockIdx.y;
  const int qs0  = (blockIdx.x * 4 + wave) * 16;

  const bf16* Qbase = Qb + ((size_t)bh * SEQ + qs0) * DH;
  const bf16* Kbase = Kb + (size_t)bh * SEQ * DH;
  const bf16* Vbase = Vt + (size_t)bh * DH * SEQ;

  Frag16 qf[2];
#pragma unroll
  for (int ks = 0; ks < 2; ++ks) {
    const bf16* ap = Qbase + (size_t)r * DH + ks * 32 + half * 8;
    qf[ks].q[0] = *(const uint4*)(ap);
    qf[ks].q[1] = *(const uint4*)(ap + 16);
  }

  float m[8], l[8];
  v8f o[4] = {};
#pragma unroll
  for (int j = 0; j < 8; ++j) { m[j] = -1e30f; l[j] = 0.0f; }
  bf16* myP = sP[wave];

  for (int kv0 = 0; kv0 < SEQ; kv0 += 32) {
    // ---- all four K fragments issued as one batch ----
    Frag16 kf[2][2];
#pragma unroll
    for (int t2 = 0; t2 < 2; ++t2)
#pragma unroll
      for (int ks = 0; ks < 2; ++ks) {
        const bf16* bp = Kbase + (size_t)(kv0 + t2 * 16 + r) * DH + ks * 32 + half * 8;
        kf[t2][ks].q[0] = *(const uint4*)(bp);
        kf[t2][ks].q[1] = *(const uint4*)(bp + 16);
      }
    v8f sc[2] = {};
#pragma unroll
    for (int t2 = 0; t2 < 2; ++t2) {
      sc[t2] = wmma_bf16(qf[0].v, kf[t2][0].v, sc[t2]);
      sc[t2] = wmma_bf16(qf[1].v, kf[t2][1].v, sc[t2]);
    }
    // ---- prefetch V fragments; they complete under the softmax VALU work ----
    Frag16 vf[4];
#pragma unroll
    for (int t = 0; t < 4; ++t) {
      const bf16* vp = Vbase + (size_t)(t * 16 + r) * SEQ + kv0 + half * 8;
      vf[t].q[0] = *(const uint4*)(vp);
      vf[t].q[1] = *(const uint4*)(vp + 16);
    }
    // ---- online softmax (row stats live across a 16-lane half) ----
#pragma unroll
    for (int j = 0; j < 8; ++j) {
      float a0 = sc[0][j] * 0.125f;      // 1/sqrt(64)
      float a1 = sc[1][j] * 0.125f;
      float mt = fmaxf(a0, a1);
      mt = fmaxf(mt, __shfl_xor(mt, 1));
      mt = fmaxf(mt, __shfl_xor(mt, 2));
      mt = fmaxf(mt, __shfl_xor(mt, 4));
      mt = fmaxf(mt, __shfl_xor(mt, 8));
      float mn = fmaxf(m[j], mt);
      float c  = __expf(m[j] - mn);
      m[j] = mn;
      float p0 = __expf(a0 - mn);
      float p1 = __expf(a1 - mn);
      float rs = p0 + p1;
      rs += __shfl_xor(rs, 1);
      rs += __shfl_xor(rs, 2);
      rs += __shfl_xor(rs, 4);
      rs += __shfl_xor(rs, 8);
      l[j] = l[j] * c + rs;
      const int row = j + half * 8;
      myP[row * 32 + r]      = (bf16)p0;
      myP[row * 32 + 16 + r] = (bf16)p1;
#pragma unroll
      for (int t = 0; t < 4; ++t) o[t][j] *= c;
    }
    asm volatile("" ::: "memory");  // keep ds stores before ds loads (same-wave LDS is in-order)
    Frag16 pf;
    {
      const bf16* lp = myP + r * 32 + half * 8;
      pf.q[0] = *(const uint4*)(lp);
      pf.q[1] = *(const uint4*)(lp + 16);
    }
#pragma unroll
    for (int t = 0; t < 4; ++t) o[t] = wmma_bf16(pf.v, vf[t].v, o[t]);
    asm volatile("" ::: "memory");
  }

  const int b = bh >> 4, h = bh & 15;
#pragma unroll
  for (int j = 0; j < 8; ++j) {
    const float inv = 1.0f / l[j];
    const int s = qs0 + j + half * 8;
    bf16* op = Ctx + ((size_t)(b * SEQ + s)) * DTOK + h * DH + r;
#pragma unroll
    for (int t = 0; t < 4; ++t) op[t * 16] = (bf16)(o[t][j] * inv);
  }
}

// ---------------- output projection (fp32 out) ----------------
__global__ __launch_bounds__(128) void k_out(const bf16* __restrict__ ctx,
                                             const bf16* __restrict__ wto,
                                             const float* __restrict__ bo,
                                             float* __restrict__ out) {
  const int wave = threadIdx.x >> 5;
  const int row0 = (blockIdx.x * 4 + wave) * 16;
  const int n0   = blockIdx.y * 64;
  v8f acc[4] = {};
#if USE_TDM
  __shared__ __align__(16) bf16 sB[2][64 * 32];
  mm_acc_tdm(ctx, wto + (size_t)n0 * DTOK, row0, DTOK, acc, &sB[0][0]);
#else
  mm_acc_db(ctx, wto, row0, n0, DTOK, acc);
#endif
  const int lane = threadIdx.x & 31;
  const int half = lane >> 4;
  const int r    = lane & 15;
#pragma unroll
  for (int t = 0; t < 4; ++t) {
    const int n = n0 + t * 16 + r;
    const float bi = bo[n];
#pragma unroll
    for (int j = 0; j < 8; ++j)
      out[(size_t)(row0 + j + half * 8) * DTOK + n] = acc[t][j] + bi;
  }
}

// ---------------- launcher ----------------
extern "C" void kernel_launch(void* const* d_in, const int* in_sizes, int n_in,
                              void* d_out, int out_size, void* d_ws, size_t ws_size,
                              hipStream_t stream) {
  const float* x  = (const float*)d_in[0];
  const float* Wq = (const float*)d_in[1];
  const float* bq = (const float*)d_in[2];
  const float* Wk = (const float*)d_in[3];
  const float* bk = (const float*)d_in[4];
  const float* Wv = (const float*)d_in[5];
  const float* bv = (const float*)d_in[6];
  const float* Wo = (const float*)d_in[7];
  const float* bo = (const float*)d_in[8];

  char* ws = (char*)d_ws;
  const size_t XB_BYTES = (size_t)BSROWS * DTOK * 2;        // 16 MB
  const size_t WT_BYTES = (size_t)4 * DTOK * DTOK * 2;      // 8 MB
  bf16* Xb  = (bf16*)(ws);
  bf16* Wt  = (bf16*)(ws + XB_BYTES);
  bf16* Qb  = (bf16*)(ws + XB_BYTES + WT_BYTES);
  bf16* Kb  = (bf16*)(ws + XB_BYTES + WT_BYTES + XB_BYTES);
  bf16* Vt  = (bf16*)(ws + XB_BYTES + WT_BYTES + 2 * XB_BYTES);
  bf16* Ctx = (bf16*)(ws + XB_BYTES + WT_BYTES + 3 * XB_BYTES);
  bf16* Wto = Wt + (size_t)3 * DTOK * DTOK;

  const int nx = BSROWS * DTOK;
  k_cvt_x<<<(nx / 4 + 255) / 256, 256, 0, stream>>>(x, Xb, nx);
  k_cvt_wt<<<dim3(DTOK / 32, DTOK / 32, 4), 256, 0, stream>>>(Wq, Wk, Wv, Wo, Wt);
  k_qkv<<<dim3(BSROWS / 64, DTOK / 64, 3), 128, 0, stream>>>(Xb, Wt, bq, bk, bv, Qb, Kb, Vt);
  k_attn<<<dim3(SEQ / 64, BATCH * NH), 128, 0, stream>>>(Qb, Kb, Vt, Ctx);
  k_out<<<dim3(BSROWS / 64, DTOK / 64), 128, 0, stream>>>(Ctx, Wto, bo, (float*)d_out);
}
